// ClusteringLayer_2405181686143
// MI455X (gfx1250) — matrix operand order repaired and checked by reading, not compiled, not run on hardware
//
#include <hip/hip_runtime.h>

// ---- types ----
typedef __attribute__((ext_vector_type(16))) __bf16 v16bf;
typedef __attribute__((ext_vector_type(8)))  __bf16 v8bf;
typedef __attribute__((ext_vector_type(4)))  __bf16 v4bf;
typedef __attribute__((ext_vector_type(2)))  __bf16 v2bf;
typedef __attribute__((ext_vector_type(8)))  float  v8f;

#define DD 512              // feature dim (reduction)
#define KK 512              // number of clusters (output cols)
#define M_TILE 64           // rows per block

union V16U { v16bf v; v8bf h[2]; };

// ------------------------------------------------------------------
// Prep: c2[k] = ||cluster_k||^2 ; split clusters into bf16 hi/lo and
// store pre-swizzled in WMMA B-matrix (32x16, K-major tiles) lane layout:
//   element (r=feature, c=cluster):
//     tile  = (r>>5)*32 + (c>>4)          (tiles ordered [rchunk][ctile])
//     lane  = ((r>>4)&1)*16 + (c&15)      (lanes 16..31 hold K=16..31)
//     i     = r&15
//     off   = tile*512 + lane*16 + i
// ------------------------------------------------------------------
__global__ void __launch_bounds__(256) prep_kernel(
    const float* __restrict__ clusters,
    __bf16* __restrict__ wh, __bf16* __restrict__ wl,
    float* __restrict__ c2) {
  __shared__ float red[256];
  const int c = blockIdx.x;      // cluster index
  const int t = threadIdx.x;     // 0..255, handles features 2t, 2t+1
  const float2 v = *(const float2*)(clusters + (size_t)c * DD + t * 2);

  const int r0   = t * 2;
  const int tile = ((r0 >> 5) * 32) + (c >> 4);
  const int lane = (((r0 >> 4) & 1) << 4) | (c & 15);
  const int off  = tile * 512 + lane * 16 + (r0 & 15);

  const __bf16 h0 = (__bf16)v.x;
  const __bf16 h1 = (__bf16)v.y;
  const __bf16 l0 = (__bf16)(v.x - (float)h0);
  const __bf16 l1 = (__bf16)(v.y - (float)h1);
  v2bf hv; hv[0] = h0; hv[1] = h1;
  v2bf lv; lv[0] = l0; lv[1] = l1;
  *(v2bf*)(wh + off) = hv;
  *(v2bf*)(wl + off) = lv;

  red[t] = v.x * v.x + v.y * v.y;
  __syncthreads();
  for (int s = 128; s > 0; s >>= 1) {
    if (t < s) red[t] += red[t + s];
    __syncthreads();
  }
  if (t == 0) c2[c] = red[0];
}

// ------------------------------------------------------------------
// Main fused kernel: 64 rows x 512 cols per block, 8 waves.
//   wave w: ALL 4 row-tiles x col-group w (4 col-tiles = 64 columns)
// bf16x3 GEMM on v_wmma_f32_16x16x32_bf16, fused 1/(1+d2) + row norm.
// ------------------------------------------------------------------
__global__ void __launch_bounds__(256) cluster_kernel(
    const float* __restrict__ x,
    const __bf16* __restrict__ bh, const __bf16* __restrict__ bl,
    const float* __restrict__ c2g,
    float* __restrict__ out) {
  __shared__ __bf16 sxh[M_TILE * DD];   // 64 KB
  __shared__ __bf16 sxl[M_TILE * DD];   // 64 KB
  __shared__ float  sx2[M_TILE];
  __shared__ float  spart[M_TILE][8];
  __shared__ float  sinv[M_TILE];

  const int tid  = threadIdx.x;
  const size_t row0 = (size_t)blockIdx.x * M_TILE;

  if (tid < M_TILE) sx2[tid] = 0.0f;
  __syncthreads();

  // ---- stage x tile: fp32 -> bf16 hi/lo into LDS, accumulate ||x||^2 ----
  {
    const float4* xb = (const float4*)(x + row0 * DD);
    for (int it = 0; it < (M_TILE * DD / 4) / 256; ++it) {
      const int j = tid + it * 256;          // float4 index
      const float4 v = xb[j];
      const __bf16 h0 = (__bf16)v.x, h1 = (__bf16)v.y,
                   h2 = (__bf16)v.z, h3 = (__bf16)v.w;
      v4bf hv; hv[0] = h0; hv[1] = h1; hv[2] = h2; hv[3] = h3;
      v4bf lv;
      lv[0] = (__bf16)(v.x - (float)h0);
      lv[1] = (__bf16)(v.y - (float)h1);
      lv[2] = (__bf16)(v.z - (float)h2);
      lv[3] = (__bf16)(v.w - (float)h3);
      *(v4bf*)(sxh + j * 4) = hv;
      *(v4bf*)(sxl + j * 4) = lv;
      atomicAdd(&sx2[j >> 7], v.x * v.x + v.y * v.y + v.z * v.z + v.w * v.w);
    }
  }
  __syncthreads();

  const int wave  = tid >> 5;
  const int lane  = tid & 31;
  const int cg    = wave;           // col group (0..7): col-tiles cg*4..cg*4+3
  const int lhalf = lane >> 4;      // 0 / 1
  const int l15   = lane & 15;
  const int kbase = lhalf * 16;     // A layout: lanes 16..31 hold K=16..31

  v8f acc[16] = {};                 // acc[rt*4 + ct]

  // ---- GEMM: 16 K-chunks of 32; B regs reused across 4 row-tiles ----
  for (int rc = 0; rc < DD; rc += 32) {
    V16U ah[4], al[4];
#pragma unroll
    for (int rt = 0; rt < 4; ++rt) {
      const int arow = rt * 16 + l15;
      ah[rt].h[0] = *(const v8bf*)(sxh + arow * DD + rc + kbase);
      ah[rt].h[1] = *(const v8bf*)(sxh + arow * DD + rc + 8 + kbase);
      al[rt].h[0] = *(const v8bf*)(sxl + arow * DD + rc + kbase);
      al[rt].h[1] = *(const v8bf*)(sxl + arow * DD + rc + 8 + kbase);
    }
    const size_t tb = ((size_t)(rc >> 5) * 32 + cg * 4) * 512 + lane * 16;

#pragma unroll
    for (int ct = 0; ct < 4; ++ct) {
      V16U bhv, blv;
      bhv.h[0] = *(const v8bf*)(bh + tb + ct * 512);
      bhv.h[1] = *(const v8bf*)(bh + tb + ct * 512 + 8);
      blv.h[0] = *(const v8bf*)(bl + tb + ct * 512);
      blv.h[1] = *(const v8bf*)(bl + tb + ct * 512 + 8);
#pragma unroll
      for (int rt = 0; rt < 4; ++rt) {
        acc[rt * 4 + ct] = __builtin_amdgcn_wmma_f32_16x16x32_bf16(
            false, ah[rt].v, false, bhv.v, (short)0, acc[rt * 4 + ct], false, false);
        acc[rt * 4 + ct] = __builtin_amdgcn_wmma_f32_16x16x32_bf16(
            false, ah[rt].v, false, blv.v, (short)0, acc[rt * 4 + ct], false, false);
        acc[rt * 4 + ct] = __builtin_amdgcn_wmma_f32_16x16x32_bf16(
            false, al[rt].v, false, bhv.v, (short)0, acc[rt * 4 + ct], false, false);
      }
    }
  }

  // ---- epilogue: q = 1/(1+d2), row partial sums per col-group ----
#pragma unroll
  for (int rt = 0; rt < 4; ++rt) {
    float x2v[8], s[8];
#pragma unroll
    for (int v = 0; v < 8; ++v) {
      x2v[v] = sx2[rt * 16 + lhalf * 8 + v];  // C layout: lanes 16..31 -> M=8+v
      s[v] = 0.0f;
    }
#pragma unroll
    for (int ct = 0; ct < 4; ++ct) {
      const float c2v = c2g[(cg * 4 + ct) * 16 + l15];
#pragma unroll
      for (int v = 0; v < 8; ++v) {
        float d = x2v[v] + c2v - 2.0f * acc[rt * 4 + ct][v];
        d = fmaxf(d, 0.0f);
        const float q = 1.0f / (1.0f + d);
        acc[rt * 4 + ct][v] = q;
        s[v] += q;
      }
    }
    // reduce across 16 lanes within each half-wave
#pragma unroll
    for (int v = 0; v < 8; ++v) {
      s[v] += __shfl_xor(s[v], 1, 32);
      s[v] += __shfl_xor(s[v], 2, 32);
      s[v] += __shfl_xor(s[v], 4, 32);
      s[v] += __shfl_xor(s[v], 8, 32);
    }
    if (l15 == 0) {
#pragma unroll
      for (int v = 0; v < 8; ++v)
        spart[rt * 16 + lhalf * 8 + v][cg] = s[v];
    }
  }
  __syncthreads();
  if (tid < M_TILE) {
    float t = 0.0f;
#pragma unroll
    for (int g = 0; g < 8; ++g) t += spart[tid][g];
    sinv[tid] = 1.0f / t;
  }
  __syncthreads();

  // ---- normalize + store ----
#pragma unroll
  for (int rt = 0; rt < 4; ++rt) {
    float invv[8];
#pragma unroll
    for (int v = 0; v < 8; ++v) invv[v] = sinv[rt * 16 + lhalf * 8 + v];
#pragma unroll
    for (int ct = 0; ct < 4; ++ct) {
      const int col = (cg * 4 + ct) * 16 + l15;
#pragma unroll
      for (int v = 0; v < 8; ++v) {
        const size_t row = row0 + rt * 16 + lhalf * 8 + v;
        out[row * KK + col] = acc[rt * 4 + ct][v] * invv[v];
      }
    }
  }
}

// ------------------------------------------------------------------
extern "C" void kernel_launch(void* const* d_in, const int* in_sizes, int n_in,
                              void* d_out, int out_size, void* d_ws, size_t ws_size,
                              hipStream_t stream) {
  (void)in_sizes; (void)n_in; (void)out_size; (void)ws_size;
  const float* x        = (const float*)d_in[0];
  const float* clusters = (const float*)d_in[1];
  float* out = (float*)d_out;

  char* ws = (char*)d_ws;
  __bf16* wh = (__bf16*)ws;                      // 512*512*2 = 524288 B
  __bf16* wl = (__bf16*)(ws + 524288);           // 524288 B
  float*  c2 = (float*)(ws + 1048576);           // 2048 B

  prep_kernel<<<KK, 256, 0, stream>>>(clusters, wh, wl, c2);

  const int nblocks = 131072 / M_TILE;           // 2048
  cluster_kernel<<<nblocks, 256, 0, stream>>>(x, wh, wl, c2, out);
}
